// KANLayer_29051158790858
// MI455X (gfx1250) — matrix operand order repaired and checked by reading, not compiled, not run on hardware
//
#include <hip/hip_runtime.h>
#include <hip/hip_bf16.h>

// Problem dims (KAN layer): B=8192, I=1024, O=1024, 8 spline coeffs + 1 silu col.
#define B_DIM 8192
#define I_DIM 1024
#define O_DIM 1024
#define NCOL  9                      // [silu | 8 quadratic B-spline basis values]
#define K_DIM (I_DIM * NCOL)         // 9216 = 288 * 32

typedef __attribute__((ext_vector_type(16))) _Float16 v16h;
typedef __attribute__((ext_vector_type(8)))  _Float16 v8h;
typedef __attribute__((ext_vector_type(8)))  float    v8f;

// ---------------------------------------------------------------------------
// Kernel 1: activations. silu(x) + 8 quadratic B-spline basis values on uniform
// knots t_j = -9 + 1.8 j (the reference's d>1e-8 guards are vacuous).
// ---------------------------------------------------------------------------
__global__ void kan_activation_kernel(const float* __restrict__ x,
                                      _Float16* __restrict__ A)
{
    int idx = blockIdx.x * blockDim.x + threadIdx.x;   // idx = b*I + i
    if (idx >= B_DIM * I_DIM) return;
    float xv = x[idx];
    float s  = xv / (1.0f + __expf(-xv));              // silu

    const float lo = -9.0f, h = 1.8f;
    float b0[10];
#pragma unroll
    for (int j = 0; j < 10; ++j) {
        float tj = lo + h * (float)j;
        b0[j] = (xv >= tj && xv < tj + h) ? 1.0f : 0.0f;
    }
    float b1[9];
    const float rh = 1.0f / h;
#pragma unroll
    for (int j = 0; j < 9; ++j) {
        float tj = lo + h * (float)j;
        b1[j] = (xv - tj) * rh * b0[j] + ((tj + 2.0f * h) - xv) * rh * b0[j + 1];
    }
    float b2[8];
    const float r2h = 0.5f / h;
#pragma unroll
    for (int j = 0; j < 8; ++j) {
        float tj = lo + h * (float)j;
        b2[j] = (xv - tj) * r2h * b1[j] + ((tj + 3.0f * h) - xv) * r2h * b1[j + 1];
    }

    _Float16* dst = A + (size_t)idx * NCOL;
    dst[0] = (_Float16)s;
#pragma unroll
    for (int j = 0; j < 8; ++j) dst[1 + j] = (_Float16)b2[j];
}

// ---------------------------------------------------------------------------
// Kernel 2: pack [base_weight | spline_weight] into one f16 K-major matrix
// W[o][i*9 + c] so base+spline become a single GEMM over K = 9216.
// ---------------------------------------------------------------------------
__global__ void kan_pack_weights(const float* __restrict__ bw,
                                 const float* __restrict__ sw,
                                 _Float16* __restrict__ W)
{
    int idx = blockIdx.x * blockDim.x + threadIdx.x;   // idx = o*I + i
    if (idx >= O_DIM * I_DIM) return;
    _Float16* dst = W + (size_t)idx * NCOL;
    dst[0] = (_Float16)bw[idx];
    const float* s = sw + (size_t)idx * 8;
#pragma unroll
    for (int c = 0; c < 8; ++c) dst[1 + c] = (_Float16)s[c];
}

// ---------------------------------------------------------------------------
// gfx1250 async copy: global -> LDS, 16 bytes per lane, tracked by ASYNCcnt.
// VDST operand = per-lane LDS byte address (addr[31:0] of the flat LDS addr).
// ---------------------------------------------------------------------------
__device__ __forceinline__ void async_copy_b128(void* lds_ptr, const void* gptr)
{
    unsigned int       lds = (unsigned int)(unsigned long long)lds_ptr;
    unsigned long long ga  = (unsigned long long)gptr;
    asm volatile("global_load_async_to_lds_b128 %0, %1, off"
                 :: "v"(lds), "v"(ga) : "memory");
}
__device__ __forceinline__ void wait_async_zero()
{
    asm volatile("s_wait_asynccnt 0x0" ::: "memory");
}

// ---------------------------------------------------------------------------
// Kernel 3: WMMA GEMM.  C[8192x1024] f32 = A[8192x9216] f16 * W[1024x9216]^T
// Block tile 128x128, 8 waves (wave32) in a 4x2 grid, wave tile 32x64 =
// 2x4 WMMA 16x16 tiles.  K staged in steps of 32 through double-buffered,
// bank-conflict-padded LDS filled with global_load_async_to_lds_b128.
// ---------------------------------------------------------------------------
#define BM 128
#define BN 128
#define BK 32
#define LDT 40                        // padded LDS row stride in halves (80B)
#define NKT (K_DIM / BK)              // 288

__global__ __launch_bounds__(256)
void kan_wmma_gemm(const _Float16* __restrict__ A,
                   const _Float16* __restrict__ W,
                   float* __restrict__ C)
{
    __shared__ _Float16 Asl[2][BM * LDT];   // 2 x 10240 B
    __shared__ _Float16 Bsl[2][BN * LDT];   // 2 x 10240 B

    const int tid  = threadIdx.x;
    const int lane = tid & 31;
    const int wave = tid >> 5;
    const int wm   = (wave >> 1) * 32;   // wave row offset in block tile (4 rows)
    const int wn   = (wave & 1) * 64;    // wave col offset in block tile (2 cols)
    const int lh   = lane & 15;
    const int hi   = lane >> 4;          // half-wave select

    const int n0 = blockIdx.x * BN;
    const int m0 = blockIdx.y * BM;

    // Per-thread staging slots: 2 chunks for A tile + 2 for B tile
    // (128 rows x 4 x 16B chunks = 512 chunks, 256 threads).
    const int r0 = tid >> 2,          kp0 = (tid & 3) * 8;
    const int r1 = (tid + 256) >> 2,  kp1 = ((tid + 256) & 3) * 8;

    const _Float16* Ab = A + (size_t)m0 * K_DIM;
    const _Float16* Wb = W + (size_t)n0 * K_DIM;

    // Issue the async fill of LDS buffer `b` with K-tile `kt`.
    auto issue_tile = [&](int kt, int b) {
        const size_t kb = (size_t)kt * BK;
        async_copy_b128(&Asl[b][r0 * LDT + kp0], Ab + (size_t)r0 * K_DIM + kb + kp0);
        async_copy_b128(&Asl[b][r1 * LDT + kp1], Ab + (size_t)r1 * K_DIM + kb + kp1);
        async_copy_b128(&Bsl[b][r0 * LDT + kp0], Wb + (size_t)r0 * K_DIM + kb + kp0);
        async_copy_b128(&Bsl[b][r1 * LDT + kp1], Wb + (size_t)r1 * K_DIM + kb + kp1);
    };

    v8f acc[2][4] = {};

    issue_tile(0, 0);
    wait_async_zero();
    __syncthreads();

    for (int kt = 0; kt < NKT; ++kt) {
        const int cur = kt & 1;
        if (kt + 1 < NKT) issue_tile(kt + 1, cur ^ 1);   // overlap copy w/ MMA

        // Fragments per the gfx1250 16-bit WMMA VGPR layouts.
        v16h af[2], bf[4];
#pragma unroll
        for (int t = 0; t < 2; ++t) {
            // A 16x32: lane<16 holds K{0..7,16..23}; lane>=16 K{8..15,24..31}
            const _Float16* ar = &Asl[cur][(wm + t * 16 + lh) * LDT + hi * 8];
            v8h aLo = *(const v8h*)(ar);
            v8h aHi = *(const v8h*)(ar + 16);
#pragma unroll
            for (int e = 0; e < 8; ++e) { af[t][e] = aLo[e]; af[t][e + 8] = aHi[e]; }
        }
#pragma unroll
        for (int t = 0; t < 4; ++t) {
            // B 32x16: lane<16 holds K{0..7,16..23}; lane>=16 K{8..15,24..31} of col lh
            const _Float16* br = &Bsl[cur][(wn + t * 16 + lh) * LDT + hi * 8];
            v8h bLo = *(const v8h*)(br);
            v8h bHi = *(const v8h*)(br + 16);
#pragma unroll
            for (int e = 0; e < 8; ++e) { bf[t][e] = bLo[e]; bf[t][e + 8] = bHi[e]; }
        }

#pragma unroll
        for (int ti = 0; ti < 2; ++ti)
#pragma unroll
            for (int tj = 0; tj < 4; ++tj)
                acc[ti][tj] = __builtin_amdgcn_wmma_f32_16x16x32_f16(
                    false, af[ti], false, bf[tj], (short)0, acc[ti][tj],
                    false, false);

        if (kt + 1 < NKT) {
            wait_async_zero();     // our async writes for kt+1 have landed
            __syncthreads();       // everyone's writes landed / reads done
        }
    }

    // Epilogue: C/D layout = VGPR r holds M=r (lanes 0-15) / M=r+8 (16-31),
    // N = lane&15.  Coalesced f32 stores per half-wave.
#pragma unroll
    for (int ti = 0; ti < 2; ++ti) {
#pragma unroll
        for (int tj = 0; tj < 4; ++tj) {
            int grow = m0 + wm + ti * 16 + hi * 8;
            int gcol = n0 + wn + tj * 16 + lh;
            float* cp = C + (size_t)grow * O_DIM + gcol;
#pragma unroll
            for (int r = 0; r < 8; ++r)
                cp[(size_t)r * O_DIM] = acc[ti][tj][r];
        }
    }
}

// ---------------------------------------------------------------------------
extern "C" void kernel_launch(void* const* d_in, const int* in_sizes, int n_in,
                              void* d_out, int out_size, void* d_ws, size_t ws_size,
                              hipStream_t stream)
{
    (void)in_sizes; (void)n_in; (void)out_size; (void)ws_size;
    const float* x  = (const float*)d_in[0];   // (B, I)    f32
    const float* bw = (const float*)d_in[1];   // (O, I)    f32
    const float* sw = (const float*)d_in[2];   // (O, I, 8) f32
    float* out = (float*)d_out;                // (B, O)    f32

    // Workspace: A (B x 9216 f16, 151 MB) then W (O x 9216 f16, 19 MB).
    _Float16* Aact = (_Float16*)d_ws;
    _Float16* Wp   = Aact + (size_t)B_DIM * K_DIM;   // byte off 150994944 (16B aligned)

    kan_activation_kernel<<<(B_DIM * I_DIM) / 256, 256, 0, stream>>>(x, Aact);
    kan_pack_weights<<<(O_DIM * I_DIM) / 256, 256, 0, stream>>>(bw, sw, Wp);

    dim3 grid(O_DIM / BN, B_DIM / BM);   // (8, 64)
    kan_wmma_gemm<<<grid, 256, 0, stream>>>(Aact, Wp, out);
}
